// Qwen3MoeSparseMoeBlock_2413771621126
// MI455X (gfx1250) — matrix operand order repaired and checked by reading, not compiled, not run on hardware
//
#include <hip/hip_runtime.h>

// ---------------- types for WMMA (gfx1250, wave32) ----------------
typedef __bf16 bf16_t;
typedef bf16_t v8bf  __attribute__((ext_vector_type(8)));
typedef bf16_t v16bf __attribute__((ext_vector_type(16)));
typedef float  v8f   __attribute__((ext_vector_type(8)));

// ---------------- problem constants ----------------
constexpr int T    = 1024;   // tokens (2*512)
constexpr int H    = 2048;   // hidden
constexpr int E    = 16;     // experts
constexpr int TOPK = 4;
constexpr int I    = 768;    // intermediate
constexpr int TWOI = 1536;

__device__ __forceinline__ unsigned short f2bf(float f) {
  unsigned u = __builtin_bit_cast(unsigned, f);
  u = (u + 0x7fffu + ((u >> 16) & 1u)) >> 16;   // round-to-nearest-even
  return (unsigned short)u;
}

__device__ __forceinline__ v16bf cat8(v8bf lo, v8bf hi) {
  v16bf r;
#pragma unroll
  for (int i = 0; i < 8; ++i) { r[i] = lo[i]; r[i + 8] = hi[i]; }
  return r;
}

// ================= 1. router: logits -> top4 -> combine weights =================
__global__ __launch_bounds__(256) void router_kernel(
    const float* __restrict__ x, const float* __restrict__ rw,
    float* __restrict__ combine) {
  __shared__ float lg[16][17];
  __shared__ int   sidx[16][TOPK];
  __shared__ float sw[16][TOPK];
  int tid = threadIdx.x;
  int e = tid & 15, tl = tid >> 4;
  int t = blockIdx.x * 16 + tl;
  const float4* xr = reinterpret_cast<const float4*>(x + (size_t)t * H);
  const float4* wr = reinterpret_cast<const float4*>(rw + (size_t)e * H);
  float acc = 0.f;
  for (int j = 0; j < H / 4; ++j) {
    float4 a = xr[j], b = wr[j];
    acc += a.x * b.x + a.y * b.y + a.z * b.z + a.w * b.w;
  }
  lg[tl][e] = acc;
  __syncthreads();
  if (e == 0) {
    float mx = lg[tl][0];
    for (int j = 1; j < 16; ++j) mx = fmaxf(mx, lg[tl][j]);
    bool taken[16];
    for (int j = 0; j < 16; ++j) taken[j] = false;
    float vals[TOPK]; int idxs[TOPK]; float s = 0.f;
    for (int k = 0; k < TOPK; ++k) {
      float best = -3.4e38f; int bi = 0;
      for (int j = 0; j < 16; ++j)
        if (!taken[j] && lg[tl][j] > best) { best = lg[tl][j]; bi = j; }
      taken[bi] = true;
      float v = __expf(best - mx);
      vals[k] = v; idxs[k] = bi; s += v;
    }
    float inv_s = 1.f / s;
    for (int k = 0; k < TOPK; ++k) { sidx[tl][k] = idxs[k]; sw[tl][k] = vals[k] * inv_s; }
  }
  __syncthreads();
  float w = 0.f;
  for (int k = 0; k < TOPK; ++k)
    if (sidx[tl][k] == e) w = sw[tl][k];
  combine[(size_t)t * E + e] = w;
}

// ================= 2. convert x to bf16 =================
__global__ __launch_bounds__(256) void cvt_x_kernel(const float* __restrict__ x,
                                                    unsigned short* __restrict__ xb) {
  size_t i = ((size_t)blockIdx.x * 256 + threadIdx.x) * 8;
  const float4* p = reinterpret_cast<const float4*>(x + i);
  float4 a = p[0], b = p[1];
  uint4 o;
  o.x = f2bf(a.x) | ((unsigned)f2bf(a.y) << 16);
  o.y = f2bf(a.z) | ((unsigned)f2bf(a.w) << 16);
  o.z = f2bf(b.x) | ((unsigned)f2bf(b.y) << 16);
  o.w = f2bf(b.z) | ((unsigned)f2bf(b.w) << 16);
  *reinterpret_cast<uint4*>(xb + i) = o;
}

// ================= 3. gather: per-expert token lists (deterministic) =================
__global__ __launch_bounds__(512) void gather_kernel(
    const float* __restrict__ combine, int* __restrict__ list,
    int* __restrict__ inv, int* __restrict__ counts, int* __restrict__ offs) {
  __shared__ int sel[16][32];
  __shared__ int soff[17];
  int tid = threadIdx.x;
  int e = tid >> 5, l = tid & 31;
  int cnt = 0;
  for (int c = 0; c < T / 32; ++c) {
    int t = c * 32 + l;
    float w = combine[(size_t)t * E + e];
    int s = (w > 0.f) ? 1 : 0;
    sel[e][l] = s;
    __syncthreads();
    int pos = cnt, tot = 0;
    for (int j = 0; j < 32; ++j) {
      int v = sel[e][j];
      if (j < l) pos += v;
      tot += v;
    }
    if (s) {
      int kr = 0;  // rank of this expert among token's selected experts
      for (int ee = 0; ee < e; ++ee) kr += (combine[(size_t)t * E + ee] > 0.f) ? 1 : 0;
      list[e * T + pos] = t * TOPK + kr;
    }
    cnt += tot;
    __syncthreads();
  }
  if (l == 0) counts[e] = cnt;
  __syncthreads();
  if (tid == 0) {
    int acc = 0;
    for (int ee = 0; ee < 16; ++ee) { soff[ee] = acc; offs[ee] = acc; acc += counts[ee]; }
    soff[16] = acc; offs[16] = acc;
  }
  __syncthreads();
  int off = soff[e];
  for (int r = l; r < cnt; r += 32) inv[list[e * T + r]] = off + r;
}

// ================= 4. gemm1: h = silu(gate)*up*combine  (bf16 WMMA) =================
// grid: (I/64, E*32), block 256 (8 waves). Each wave: one 16x16 gate tile + matching up tile.
__global__ __launch_bounds__(256) void gemm1_kernel(
    const unsigned short* __restrict__ xb, const float* __restrict__ gup,
    const float* __restrict__ combine, const int* __restrict__ list,
    const int* __restrict__ counts, const int* __restrict__ offs,
    unsigned short* __restrict__ hsel) {
  int e = blockIdx.y >> 5;
  int rt = blockIdx.y & 31;
  int cnt = counts[e];
  if (rt * 32 >= cnt) return;
  int i0 = blockIdx.x * 64;

  __shared__ unsigned int ldsB[128 * 20];  // 128 cols (64 gate + 64 up) x 32 k, bf16, stride 80B

  int tid = threadIdx.x;
  int lane = tid & 31, wv = tid >> 5;
  int half = lane >> 4, n = lane & 15;
  int mt = wv >> 2, nt = wv & 3;

  // A row (gathered token)
  int rloc = rt * 32 + mt * 16 + n;  // n doubles as M index for A loads
  int entry = (rloc < cnt) ? list[e * T + rloc] : 0;
  int t = entry >> 2;
  const unsigned short* aptr = xb + (size_t)t * H + half * 16;

  const unsigned int* bg = ldsB + (nt * 16 + n) * 20 + half * 8;
  const unsigned int* bu = ldsB + (64 + nt * 16 + n) * 20 + half * 8;

  const float* wbase = gup + (size_t)e * H * TWOI;

  v8f cg = {}, cu = {};
  for (int s = 0; s < H / 32; ++s) {
    int k0 = s * 32;
    float f0[8], f1[8];
#pragma unroll
    for (int it = 0; it < 8; ++it) {
      int idx = tid + it * 256;           // 0..2047
      int cl = idx & 127, kp = idx >> 7;  // col-local, k-pair
      int gcol = (cl < 64) ? (i0 + cl) : (I + i0 + (cl - 64));
      const float* p = wbase + (size_t)(k0 + 2 * kp) * TWOI + gcol;
      f0[it] = p[0];
      f1[it] = p[TWOI];
      if (s + 1 < H / 32) __builtin_prefetch(p + 32 * TWOI, 0, 1);
    }
    __syncthreads();  // previous iteration's LDS reads done
#pragma unroll
    for (int it = 0; it < 8; ++it) {
      int idx = tid + it * 256;
      int cl = idx & 127, kp = idx >> 7;
      ldsB[cl * 20 + kp] = (unsigned)f2bf(f0[it]) | ((unsigned)f2bf(f1[it]) << 16);
    }
    __syncthreads();

    v8bf alo = *reinterpret_cast<const v8bf*>(aptr + k0);
    v8bf ahi = *reinterpret_cast<const v8bf*>(aptr + k0 + 8);
    v16bf a = cat8(alo, ahi);
    v8bf g0 = *reinterpret_cast<const v8bf*>(bg);
    v8bf g1 = *reinterpret_cast<const v8bf*>(bg + 4);
    v8bf u0 = *reinterpret_cast<const v8bf*>(bu);
    v8bf u1 = *reinterpret_cast<const v8bf*>(bu + 4);
    v16bf bgf = cat8(g0, g1);
    v16bf buf = cat8(u0, u1);
    cg = __builtin_amdgcn_wmma_f32_16x16x32_bf16(false, a, false, bgf, (short)0, cg, false, false);
    cu = __builtin_amdgcn_wmma_f32_16x16x32_bf16(false, a, false, buf, (short)0, cu, false, false);
  }

  int off_e = offs[e];
  int icol = i0 + nt * 16 + n;
#pragma unroll
  for (int r = 0; r < 8; ++r) {
    int rl = rt * 32 + mt * 16 + half * 8 + r;  // C-tile row
    if (rl < cnt) {
      int en = list[e * T + rl];
      int tt = en >> 2;
      float wgt = combine[(size_t)tt * E + e];
      float g = cg[r], u = cu[r];
      float hv = (g / (1.f + __expf(-g))) * u * wgt;
      hsel[(size_t)(off_e + rl) * I + icol] = f2bf(hv);
    }
  }
}

// ================= 5. gemm2: y = h_sel @ down_e  (bf16 WMMA, f32 out) =================
// grid: (H/64, E*32), block 256.
__global__ __launch_bounds__(256) void gemm2_kernel(
    const unsigned short* __restrict__ hsel, const float* __restrict__ down,
    const int* __restrict__ counts, const int* __restrict__ offs,
    float* __restrict__ y) {
  int e = blockIdx.y >> 5;
  int rt = blockIdx.y & 31;
  int cnt = counts[e];
  if (rt * 32 >= cnt) return;
  int off_e = offs[e];
  int n0 = blockIdx.x * 64;

  __shared__ unsigned int ldsB[64 * 20];

  int tid = threadIdx.x;
  int lane = tid & 31, wv = tid >> 5;
  int half = lane >> 4, n = lane & 15;
  int mt = wv >> 2, nt = wv & 3;

  int rloc = rt * 32 + mt * 16 + n;
  int grow = off_e + ((rloc < cnt) ? rloc : 0);
  const unsigned short* aptr = hsel + (size_t)grow * I + half * 16;
  const float* wbase = down + (size_t)e * I * H;

  const unsigned int* bp = ldsB + (nt * 16 + n) * 20 + half * 8;

  v8f c = {};
  for (int s = 0; s < I / 32; ++s) {
    int k0 = s * 32;
    float f0[4], f1[4];
#pragma unroll
    for (int it = 0; it < 4; ++it) {
      int idx = tid + it * 256;          // 0..1023
      int cl = idx & 63, kp = idx >> 6;  // col-local, k-pair
      const float* p = wbase + (size_t)(k0 + 2 * kp) * H + (n0 + cl);
      f0[it] = p[0];
      f1[it] = p[H];
      if (s + 1 < I / 32) __builtin_prefetch(p + 32 * H, 0, 1);
    }
    __syncthreads();
#pragma unroll
    for (int it = 0; it < 4; ++it) {
      int idx = tid + it * 256;
      int cl = idx & 63, kp = idx >> 6;
      ldsB[cl * 20 + kp] = (unsigned)f2bf(f0[it]) | ((unsigned)f2bf(f1[it]) << 16);
    }
    __syncthreads();

    v8bf alo = *reinterpret_cast<const v8bf*>(aptr + k0);
    v8bf ahi = *reinterpret_cast<const v8bf*>(aptr + k0 + 8);
    v16bf a = cat8(alo, ahi);
    v8bf b0 = *reinterpret_cast<const v8bf*>(bp);
    v8bf b1 = *reinterpret_cast<const v8bf*>(bp + 4);
    v16bf b = cat8(b0, b1);
    c = __builtin_amdgcn_wmma_f32_16x16x32_bf16(false, a, false, b, (short)0, c, false, false);
  }

  int hcol = n0 + nt * 16 + n;
#pragma unroll
  for (int r = 0; r < 8; ++r) {
    int rl = rt * 32 + mt * 16 + half * 8 + r;
    if (rl < cnt) y[(size_t)(off_e + rl) * H + hcol] = c[r];
  }
}

// ================= 6. reduce: out[t] = sum_k y[inv[t*4+k]] =================
__global__ __launch_bounds__(256) void reduce_kernel(const float* __restrict__ y,
                                                     const int* __restrict__ inv,
                                                     float* __restrict__ out) {
  int t = blockIdx.x;
  int r0 = inv[t * TOPK + 0], r1 = inv[t * TOPK + 1];
  int r2 = inv[t * TOPK + 2], r3 = inv[t * TOPK + 3];
  const float4* y4 = reinterpret_cast<const float4*>(y);
  float4* o4 = reinterpret_cast<float4*>(out);
#pragma unroll
  for (int j = 0; j < 2; ++j) {
    int c = threadIdx.x + j * 256;  // 0..511
    float4 a = y4[(size_t)r0 * (H / 4) + c];
    float4 b = y4[(size_t)r1 * (H / 4) + c];
    float4 d = y4[(size_t)r2 * (H / 4) + c];
    float4 e = y4[(size_t)r3 * (H / 4) + c];
    float4 s;
    s.x = a.x + b.x + d.x + e.x;
    s.y = a.y + b.y + d.y + e.y;
    s.z = a.z + b.z + d.z + e.z;
    s.w = a.w + b.w + d.w + e.w;
    o4[(size_t)t * (H / 4) + c] = s;
  }
}

// ================= launch =================
extern "C" void kernel_launch(void* const* d_in, const int* in_sizes, int n_in,
                              void* d_out, int out_size, void* d_ws, size_t ws_size,
                              hipStream_t stream) {
  const float* x    = (const float*)d_in[0];   // [2,512,2048]
  const float* rw   = (const float*)d_in[1];   // [16,2048]
  const float* gup  = (const float*)d_in[2];   // [16,2048,1536]
  const float* down = (const float*)d_in[3];   // [16,768,2048]
  float* out = (float*)d_out;

  char* ws = (char*)d_ws;
  float*          combine = (float*)(ws + 0);                      // 64 KB
  int*            list    = (int*)(ws + 65536);                    // 64 KB
  int*            inv     = (int*)(ws + 131072);                   // 16 KB
  int*            counts  = (int*)(ws + 147456);                   // 64 B
  int*            offs    = (int*)(ws + 147456 + 64);              // 68 B
  unsigned short* xb      = (unsigned short*)(ws + 262144);        // 4 MB
  unsigned short* hsel    = (unsigned short*)(ws + 262144 + 4194304);            // 6 MB
  float*          y       = (float*)(ws + 262144 + 4194304 + 6291456);           // 32 MB

  router_kernel<<<dim3(T / 16), dim3(256), 0, stream>>>(x, rw, combine);
  cvt_x_kernel<<<dim3((T * H) / (256 * 8)), dim3(256), 0, stream>>>(x, xb);
  gather_kernel<<<dim3(1), dim3(512), 0, stream>>>(combine, list, inv, counts, offs);
  gemm1_kernel<<<dim3(I / 64, E * 32), dim3(256), 0, stream>>>(xb, gup, combine, list, counts, offs, hsel);
  gemm2_kernel<<<dim3(H / 64, E * 32), dim3(256), 0, stream>>>(hsel, down, counts, offs, y);
  reduce_kernel<<<dim3(T), dim3(256), 0, stream>>>(y, inv, out);
}
